// SUE_25383256719527
// MI455X (gfx1250) — matrix operand order repaired and checked by reading, not compiled, not run on hardware
//
#include <hip/hip_runtime.h>
#include <hip/hip_bf16.h>
#include <math.h>

// ---- problem constants (from reference setup_inputs) ----
#define BSZ    64
#define HH     100
#define DD     400
#define AA     200
#define CATN   19
#define NODES  118   // H + CAT-1
#define NODESP 128   // padded node count
#define KP     448   // padded K for D=400 GEMMs (mult of 64)
#define NC     16    // candidates
#define INV_ATT_SCALE 0.070710678118654752f   // 1/sqrt(200)

typedef __attribute__((ext_vector_type(16))) __bf16 v16bf;
typedef __attribute__((ext_vector_type(8)))  float  v8f;

union FragU { v16bf v; unsigned int u[8]; };

// pack two fp32 -> packed bf16 pair (lo in [15:0], hi in [31:16])
__device__ __forceinline__ unsigned int pack_bf16(float lo, float hi) {
    union { __bf16 b[2]; unsigned int u; } t;
    t.b[0] = (__bf16)lo;
    t.b[1] = (__bf16)hi;
    return t.u;
}

// =====================================================================
// bf16-WMMA GEMM, fully regular:  C[M,N] = epi(A[M,K'] @ B[K',N])
// Preconditions (guaranteed by caller via padding):
//   gridDim.x*64 == M, gridDim.y*16 == N, K % 64 == 0,
//   A base 16B-aligned, lda % 4 == 0, A rows may over-read up to 64
//   floats past K (multiplied by zero B rows), B has exactly K rows.
// 128 threads = 4 waves; 64x16 C tile per block (16x16 per wave).
// K-step 64: two v_wmma per barrier pair; LDS holds packed bf16 pairs.
// =====================================================================
__global__ __launch_bounds__(128)
void gemm_wmma_bf16(const float* __restrict__ A, long long sAb, int lda,
                    const float* __restrict__ Bm, long long sBb, int ldb,
                    float* __restrict__ C, long long sCb, int ldc,
                    const float* __restrict__ bias,
                    const float* __restrict__ R, long long sRb, int ldr,
                    int K, int relu)
{
    __shared__ __attribute__((aligned(16))) unsigned int AsP[64][36]; // 64x64 bf16, 144B stride
    __shared__ __attribute__((aligned(16))) unsigned int BsP[16][36]; // 64x16 bf16, col-major pairs

    const int tid  = threadIdx.x;
    const int wave = tid >> 5;
    const int lane = tid & 31;
    const int half = lane >> 4;
    const int col  = lane & 15;
    const int m0 = blockIdx.x * 64;
    const int n0 = blockIdx.y * 16;
    const int b  = blockIdx.z;

    const float* Ab = A  + (long long)b * sAb;
    const float* Bb = Bm + (long long)b * sBb;
    float*       Cb = C  + (long long)b * sCb;

    v8f acc = {};

    for (int k0 = 0; k0 < K; k0 += 64) {
        // A tile 64x64: 8 x float4 per thread, unconditional, coalesced
        #pragma unroll
        for (int i = 0; i < 8; ++i) {
            int e   = tid + i * 128;      // 0..1023 float4 slots
            int row = e >> 4;             // 16 float4 per row
            int c4  = e & 15;
            const float4 v = *reinterpret_cast<const float4*>(
                Ab + (long long)(m0 + row) * lda + k0 + (c4 << 2));
            AsP[row][c4 * 2]     = pack_bf16(v.x, v.y);
            AsP[row][c4 * 2 + 1] = pack_bf16(v.z, v.w);
        }
        // B tile 64x16: K-pairs packed, stored column-major
        #pragma unroll
        for (int i = 0; i < 4; ++i) {
            int e  = tid + i * 128;       // 0..511 pairs
            int kp = e >> 4;              // 32 k-pairs
            int c  = e & 15;
            long long base = (long long)(k0 + kp * 2) * ldb + n0 + c;
            BsP[c][kp] = pack_bf16(Bb[base], Bb[base + ldb]);
        }
        __syncthreads();

        const int mr = wave * 16 + col;
        #pragma unroll
        for (int s = 0; s < 2; ++s) {     // two K=32 slabs
            const int koff = s * 16;      // uint (pair) offset
            FragU fa, fb;
            #pragma unroll
            for (int i = 0; i < 8; ++i) {
                int ia = (i < 4 ? i : 8 + (i - 4)) + (half ? 4 : 0) + koff;
                fa.u[i] = AsP[mr][ia];
                fb.u[i] = BsP[col][ia];
            }
            acc = __builtin_amdgcn_wmma_f32_16x16x32_bf16(
                      false, fa.v, false, fb.v, (short)0, acc, false, false);
        }
        __syncthreads();
    }

    const float* Rb = R ? (R + (long long)b * sRb) : (const float*)nullptr;
    #pragma unroll
    for (int j = 0; j < 8; ++j) {
        int m = m0 + wave * 16 + j + half * 8;   // C/D layout: VGPR j -> M=j(+8)
        int n = n0 + col;
        float v = acc[j];
        if (bias) v += bias[n];
        if (relu) v = fmaxf(v, 0.f);
        if (Rb)   v += Rb[(long long)m * ldr + n];
        Cb[(long long)m * ldc + n] = v;
    }
}

// =====================================================================
// he = concat(history, proxy, zero-pad) over 128 node rows ; h = he
// =====================================================================
__global__ void build_he(const float* __restrict__ hist,
                         const float* __restrict__ proxy,
                         float* __restrict__ he, float* __restrict__ h)
{
    long long idx = (long long)blockIdx.x * blockDim.x + threadIdx.x;
    long long total = (long long)BSZ * NODESP * DD;
    if (idx >= total) return;
    int d = (int)(idx % DD);
    int r = (int)((idx / DD) % NODESP);
    long long b = idx / ((long long)DD * NODESP);
    float v = 0.f;
    if (r < HH)          v = hist[(b * HH + r) * DD + d];
    else if (r < NODES)  v = proxy[(long long)(r - HH) * DD + d];
    he[idx] = v;
    h[idx]  = v;
}

// =====================================================================
// graphP[b][128][128] = zero-padded user_history_graph[b][118][118]
// =====================================================================
__global__ void pad_graph(const float* __restrict__ g, float* __restrict__ gp)
{
    long long idx = (long long)blockIdx.x * blockDim.x + threadIdx.x;
    long long total = (long long)BSZ * NODESP * NODESP;
    if (idx >= total) return;
    int c = (int)(idx % NODESP);
    int r = (int)((idx / NODESP) % NODESP);
    long long b = idx / ((long long)NODESP * NODESP);
    gp[idx] = (r < NODES && c < NODES)
                ? g[(b * NODES + r) * NODES + c] : 0.f;
}

// =====================================================================
// dst[Kp][Ncols] = src[K][Ncols] with zero rows K..Kp-1
// =====================================================================
__global__ void pad_rows(const float* __restrict__ src, float* __restrict__ dst,
                         int K, int Kp, int Ncols)
{
    long long idx = (long long)blockIdx.x * blockDim.x + threadIdx.x;
    long long total = (long long)Kp * Ncols;
    if (idx >= total) return;
    int k = (int)(idx / Ncols);
    dst[idx] = (k < K) ? src[idx] : 0.f;
}

// =====================================================================
// candP = cand (plus zeroed slack words for safe K over-read)
// =====================================================================
__global__ void copy_cand(const float* __restrict__ src, float* __restrict__ dst)
{
    long long idx = (long long)blockIdx.x * blockDim.x + threadIdx.x;
    long long total = (long long)BSZ * NC * DD + 64;   // +slack
    if (idx >= total) return;
    dst[idx] = (idx < (long long)BSZ * NC * DD) ? src[idx] : 0.f;
}

// =====================================================================
// row-wise LayerNorm over D=400:  h = LN(z)*scale + bias
// =====================================================================
__global__ __launch_bounds__(128)
void layernorm_k(const float* __restrict__ z,
                 const float* __restrict__ scale,
                 const float* __restrict__ bias,
                 float* __restrict__ h)
{
    long long row = blockIdx.x;
    const float* zr = z + row * DD;
    float*       hr = h + row * DD;
    __shared__ float r1[128], r2[128];
    float s = 0.f, s2 = 0.f;
    for (int d = threadIdx.x; d < DD; d += 128) {
        float v = zr[d]; s += v; s2 += v * v;
    }
    r1[threadIdx.x] = s; r2[threadIdx.x] = s2;
    __syncthreads();
    for (int off = 64; off > 0; off >>= 1) {
        if (threadIdx.x < off) {
            r1[threadIdx.x] += r1[threadIdx.x + off];
            r2[threadIdx.x] += r2[threadIdx.x + off];
        }
        __syncthreads();
    }
    float mu  = r1[0] * (1.f / DD);
    float var = r2[0] * (1.f / DD) - mu * mu;
    float inv = rsqrtf(var + 1e-5f);
    for (int d = threadIdx.x; d < DD; d += 128)
        hr[d] = (zr[d] - mu) * inv * scale[d] + bias[d];
}

// =====================================================================
// gcn_feat = (h + he)[:, :H]
// =====================================================================
__global__ void add_feat(const float* __restrict__ h,
                         const float* __restrict__ he,
                         float* __restrict__ gf)
{
    long long idx = (long long)blockIdx.x * blockDim.x + threadIdx.x;
    long long total = (long long)BSZ * HH * DD;
    if (idx >= total) return;
    int d = (int)(idx % DD);
    int r = (int)((idx / DD) % HH);
    long long b = idx / ((long long)DD * HH);
    long long src = (b * NODESP + r) * DD + d;
    gf[idx] = h[src] + he[src];
}

// =====================================================================
// Per-(b,n): attention logits, segment(softmax by category), scatter-sum
// into 19x400 cluster features (accumulated in LDS).
// =====================================================================
__global__ __launch_bounds__(128)
void intra_kernel(const float* __restrict__ Kbuf,   // [B,H,A]
                  const float* __restrict__ Qbuf,   // [B,N,A]
                  const float* __restrict__ gf,     // [B,H,D]
                  const int*   __restrict__ catidx, // [B,H]
                  float* __restrict__ intra)        // [B,N,C,D]
{
    const int n = blockIdx.x, b = blockIdx.y;
    __shared__ float q_sh[AA];
    __shared__ float a_sh[HH];
    __shared__ int   cat_sh[HH];
    __shared__ float cmax[CATN], csum[CATN];
    __shared__ float accum[CATN * DD];   // 19*400 = 7600 floats (~30KB)
    const int tid = threadIdx.x;

    for (int i = tid; i < AA; i += 128)
        q_sh[i] = Qbuf[((long long)b * NC + n) * AA + i];
    for (int i = tid; i < HH; i += 128)
        cat_sh[i] = catidx[b * HH + i];
    __syncthreads();

    if (tid < HH) {
        const float* kr = Kbuf + ((long long)b * HH + tid) * AA;
        float s = 0.f;
        for (int a = 0; a < AA; ++a) s += kr[a] * q_sh[a];
        a_sh[tid] = s * INV_ATT_SCALE;
    }
    __syncthreads();

    if (tid < CATN) {
        float mx = -3.4e38f;
        for (int hh = 0; hh < HH; ++hh)
            if (cat_sh[hh] == tid) mx = fmaxf(mx, a_sh[hh]);
        float sm = 0.f;
        for (int hh = 0; hh < HH; ++hh)
            if (cat_sh[hh] == tid) sm += __expf(a_sh[hh] - mx);
        cmax[tid] = mx; csum[tid] = sm;
    }
    __syncthreads();

    if (tid < HH) {
        int c = cat_sh[tid];
        a_sh[tid] = __expf(a_sh[tid] - cmax[c]) / csum[c];   // alpha
    }
    for (int i = tid; i < CATN * DD; i += 128) accum[i] = 0.f;
    __syncthreads();

    // scatter: each thread owns distinct d columns -> no LDS write races
    for (int d = tid; d < DD; d += 128) {
        for (int hh = 0; hh < HH; ++hh) {
            accum[cat_sh[hh] * DD + d] +=
                a_sh[hh] * gf[((long long)b * HH + hh) * DD + d];
        }
    }
    __syncthreads();

    float* op = intra + ((long long)b * NC + n) * CATN * DD;
    for (int i = tid; i < CATN * DD; i += 128) op[i] = accum[i];
}

// =====================================================================
// Per-(b,n): inter-cluster masked softmax attention over 19 clusters.
// =====================================================================
__global__ __launch_bounds__(128)
void inter_kernel(const float* __restrict__ Kf,     // [B,N,C,A]
                  const float* __restrict__ qic,    // [B,N,A]
                  const float* __restrict__ intra2, // [B,N,C,D]
                  const int*   __restrict__ mask,   // [B,C]
                  float* __restrict__ out)          // [B,N,D]
{
    const int n = blockIdx.x, b = blockIdx.y;
    __shared__ float q_sh[AA];
    __shared__ float w_sh[CATN];
    const int tid = threadIdx.x;

    for (int i = tid; i < AA; i += 128)
        q_sh[i] = qic[((long long)b * NC + n) * AA + i];
    __syncthreads();

    if (tid < CATN) {
        const float* kr = Kf + (((long long)b * NC + n) * CATN + tid) * AA;
        float s = 0.f;
        for (int a = 0; a < AA; ++a) s += kr[a] * q_sh[a];
        int mk = (tid == CATN - 1) ? 1 : mask[b * CATN + tid];
        w_sh[tid] = mk ? (s * INV_ATT_SCALE) : -3.4e38f;
    }
    __syncthreads();

    if (tid == 0) {
        float mx = -3.4e38f;
        for (int c = 0; c < CATN; ++c) mx = fmaxf(mx, w_sh[c]);
        float sm = 0.f;
        for (int c = 0; c < CATN; ++c) { float e = __expf(w_sh[c] - mx); w_sh[c] = e; sm += e; }
        float inv = 1.f / sm;
        for (int c = 0; c < CATN; ++c) w_sh[c] *= inv;
    }
    __syncthreads();

    const float* ib = intra2 + ((long long)b * NC + n) * CATN * DD;
    for (int d = tid; d < DD; d += 128) {
        float s = 0.f;
        for (int c = 0; c < CATN; ++c) s += w_sh[c] * ib[c * DD + d];
        out[((long long)b * NC + n) * DD + d] = s;
    }
}

// =====================================================================
extern "C" void kernel_launch(void* const* d_in, const int* in_sizes, int n_in,
                              void* d_out, int out_size, void* d_ws, size_t ws_size,
                              hipStream_t stream)
{
    (void)in_sizes; (void)n_in; (void)out_size; (void)ws_size;
    const float* hist  = (const float*)d_in[0];
    const float* cand  = (const float*)d_in[1];
    const float* graph = (const float*)d_in[2];
    const int*   cmask = (const int*)  d_in[3];
    const int*   cidx  = (const int*)  d_in[4];
    const float* proxy = (const float*)d_in[5];
    const float* gcnW  = (const float*)d_in[6];
    const float* gcnb  = (const float*)d_in[7];
    const float* lnS   = (const float*)d_in[8];
    const float* lnB   = (const float*)d_in[9];
    const float* Kw    = (const float*)d_in[10];
    const float* Qw    = (const float*)d_in[11];
    const float* Qb    = (const float*)d_in[12];
    const float* affW  = (const float*)d_in[13];
    const float* affb  = (const float*)d_in[14];
    const float* icKw  = (const float*)d_in[15];
    const float* icQw  = (const float*)d_in[16];
    const float* icQb  = (const float*)d_in[17];
    float* out = (float*)d_out;

    // ---- workspace layout (floats); every A-operand region is followed
    // ---- by more allocated ws (safe 48-float K over-read) ----
    float* ws = (float*)d_ws;
    long long off = 0;
    float* gf     = ws + off; off += (long long)BSZ * HH * DD;        // 2,560,000
    float* Kbuf   = ws + off; off += (long long)BSZ * HH * AA;        // 1,280,000
    float* Qbuf   = ws + off; off += (long long)BSZ * NC * AA;        //   204,800
    float* qic    = ws + off; off += (long long)BSZ * NC * AA;        //   204,800
    float* candP  = ws + off; off += (long long)BSZ * NC * DD + 64;   //   409,664
    float* graphP = ws + off; off += (long long)BSZ * NODESP * NODESP;// 1,048,576
    float* gW0P   = ws + off; off += (long long)KP * DD;              //   179,200
    float* gW1P   = ws + off; off += (long long)KP * DD;
    float* affWP  = ws + off; off += (long long)KP * DD;
    float* KwP    = ws + off; off += (long long)KP * AA;              //    89,600
    float* QwP    = ws + off; off += (long long)KP * AA;
    float* icKwP  = ws + off; off += (long long)KP * AA;
    float* icQwP  = ws + off; off += (long long)KP * AA;
    float* pool   = ws + off;
    const long long nodeSz  = (long long)BSZ * NODESP * DD;           // 3,276,800
    const long long intraSz = (long long)BSZ * NC * CATN * DD;        // 7,782,400
    float* he   = pool;                    // GCN phase
    float* h    = pool + nodeSz;
    float* tmp  = pool + 2 * nodeSz;
    float* z    = pool + 3 * nodeSz;
    float* intra  = pool;                  // reuse after GCN phase is dead
    float* intra2 = pool + intraSz;
    float* Kf     = pool + 2 * intraSz;

    const dim3 blk(128);

    // 0) padding / staging
    {
        long long t1 = (long long)BSZ * NODESP * DD;
        build_he<<<dim3((unsigned)((t1 + 255) / 256)), dim3(256), 0, stream>>>(
            hist, proxy, he, h);
        long long t2 = (long long)BSZ * NODESP * NODESP;
        pad_graph<<<dim3((unsigned)((t2 + 255) / 256)), dim3(256), 0, stream>>>(
            graph, graphP);
        long long tw = (long long)KP * DD;
        pad_rows<<<dim3((unsigned)((tw + 255) / 256)), dim3(256), 0, stream>>>(
            gcnW, gW0P, DD, KP, DD);
        pad_rows<<<dim3((unsigned)((tw + 255) / 256)), dim3(256), 0, stream>>>(
            gcnW + (long long)DD * DD, gW1P, DD, KP, DD);
        pad_rows<<<dim3((unsigned)((tw + 255) / 256)), dim3(256), 0, stream>>>(
            affW, affWP, DD, KP, DD);
        long long ta = (long long)KP * AA;
        pad_rows<<<dim3((unsigned)((ta + 255) / 256)), dim3(256), 0, stream>>>(
            Kw, KwP, DD, KP, AA);
        pad_rows<<<dim3((unsigned)((ta + 255) / 256)), dim3(256), 0, stream>>>(
            Qw, QwP, DD, KP, AA);
        pad_rows<<<dim3((unsigned)((ta + 255) / 256)), dim3(256), 0, stream>>>(
            icKw, icKwP, DD, KP, AA);
        pad_rows<<<dim3((unsigned)((ta + 255) / 256)), dim3(256), 0, stream>>>(
            icQw, icQwP, DD, KP, AA);
        long long tc = (long long)BSZ * NC * DD + 64;
        copy_cand<<<dim3((unsigned)((tc + 255) / 256)), dim3(256), 0, stream>>>(
            cand, candP);
    }

    // 1) two GCN layers
    const float* gWP[2] = { gW0P, gW1P };
    for (int l = 0; l < 2; ++l) {
        // tmp = Gpad @ h   (batched; M=128, N=400, K=128 — all regular)
        gemm_wmma_bf16<<<dim3(NODESP / 64, DD / 16, BSZ), blk, 0, stream>>>(
            graphP, (long long)NODESP * NODESP, NODESP,
            h,      (long long)NODESP * DD,     DD,
            tmp,    (long long)NODESP * DD,     DD,
            nullptr, nullptr, 0, 0,
            NODESP, 0);
        // z = relu(tmp @ Wpad_l + b_l) + h   (M = B*128 = 8192, K = 448)
        gemm_wmma_bf16<<<dim3(BSZ * NODESP / 64, DD / 16, 1), blk, 0, stream>>>(
            tmp, 0, DD,
            gWP[l], 0, DD,
            z,   0, DD,
            gcnb + l * DD,
            h, 0, DD,
            KP, 1);
        // h = LayerNorm(z)   (pad rows harmless: killed by zero graph cols)
        layernorm_k<<<dim3(BSZ * NODESP), blk, 0, stream>>>(
            z, lnS + l * DD, lnB + l * DD, h);
    }

    // 2) gcn_feat = (h + he)[:, :H]
    {
        long long total = (long long)BSZ * HH * DD;
        add_feat<<<dim3((unsigned)((total + 255) / 256)), dim3(256), 0, stream>>>(h, he, gf);
    }

    // 3) K = gcn_feat @ Kw        (M = 6400, K = 448)
    gemm_wmma_bf16<<<dim3(BSZ * HH / 64, AA / 16, 1), blk, 0, stream>>>(
        gf, 0, DD, KwP, 0, AA, Kbuf, 0, AA,
        nullptr, nullptr, 0, 0, KP, 0);

    // 4) Q = cand @ Qw + Qb       (M = 1024, K = 448)
    gemm_wmma_bf16<<<dim3(BSZ * NC / 64, AA / 16, 1), blk, 0, stream>>>(
        candP, 0, DD, QwP, 0, AA, Qbuf, 0, AA,
        Qb, nullptr, 0, 0, KP, 0);

    // 5) segment softmax + category scatter -> intra [B,N,C,D]
    intra_kernel<<<dim3(NC, BSZ), blk, 0, stream>>>(Kbuf, Qbuf, gf, cidx, intra);

    // 6) intra2 = relu(intra @ affW + affb) + intra   (M = 19456, K = 448)
    gemm_wmma_bf16<<<dim3(BSZ * NC * CATN / 64, DD / 16, 1), blk, 0, stream>>>(
        intra, 0, DD, affWP, 0, DD, intra2, 0, DD,
        affb, intra, 0, DD, KP, 1);

    // 7) Kf = intra2 @ icKw       (M = 19456, K = 448)
    gemm_wmma_bf16<<<dim3(BSZ * NC * CATN / 64, AA / 16, 1), blk, 0, stream>>>(
        intra2, 0, DD, icKwP, 0, AA, Kf, 0, AA,
        nullptr, nullptr, 0, 0, KP, 0);

    // 8) qic = cand @ icQw + icQb (M = 1024, K = 448)
    gemm_wmma_bf16<<<dim3(BSZ * NC / 64, AA / 16, 1), blk, 0, stream>>>(
        candP, 0, DD, icQwP, 0, AA, qic, 0, AA,
        icQb, nullptr, 0, 0, KP, 0);

    // 9) masked inter-cluster attention -> out [B,N,D]
    inter_kernel<<<dim3(NC, BSZ), blk, 0, stream>>>(Kf, qic, intra2, cmask, out);
}
